// FuncGNN_82403242541727
// MI455X (gfx1250) — matrix operand order repaired and checked by reading, not compiled, not run on hardware
//
#include <hip/hip_runtime.h>

typedef __bf16 bf16;
typedef __attribute__((ext_vector_type(16))) __bf16 v16bf;
typedef __attribute__((ext_vector_type(8)))  float  v8f;

typedef int v4i __attribute__((vector_size(16)));
typedef __attribute__((address_space(1))) v4i as1_v4i;
typedef __attribute__((address_space(3))) v4i as3_v4i;

#if defined(__has_builtin)
#if __has_builtin(__builtin_amdgcn_global_load_async_to_lds_b128) && \
    __has_builtin(__builtin_amdgcn_s_wait_asynccnt)
#define USE_ASYNC_LDS 1
#endif
#endif
#ifndef USE_ASYNC_LDS
#define USE_ASYNC_LDS 0
#endif

// ---------------------------------------------------------------------------
// deterministic dropout hash: keep-prob 0.9, per (layer, node*128+feat)
// ---------------------------------------------------------------------------
__device__ __forceinline__ float drop_scale(unsigned layer, unsigned idx) {
  unsigned x = idx * 2654435761u + layer * 0x9E3779B9u;
  x ^= x >> 16; x *= 0x7feb352du;
  x ^= x >> 15; x *= 0x846ca68bu;
  x ^= x >> 16;
  return (x & 0x00FFFFFFu) < 15099494u ? (1.0f / 0.9f) : 0.0f; // 0.9*2^24
}

// ---------------------------------------------------------------------------
// Pack f32 weights [T][128][N] into bf16 WMMA B-operand tiles:
// layout [T][kc=4][nt=N/16][lane=32][16]; B 32x16 per (kc,nt):
//   lanes 0-15 hold K=0..15 (2/VGPR), lanes 16-31 hold K=16..31
// ---------------------------------------------------------------------------
__global__ void pack_w_kernel(const float* __restrict__ W, bf16* __restrict__ out,
                              int T, int NT) {
  int idx = blockIdx.x * blockDim.x + threadIdx.x;
  int total = T * 4 * NT * 512;
  if (idx >= total) return;
  int j    = idx & 15;
  int lane = (idx >> 4) & 31;
  int rest = idx >> 9;
  int nt   = rest % NT;
  int rest2 = rest / NT;
  int kc   = rest2 & 3;
  int t    = rest2 >> 2;
  int N = NT * 16;
  int n = (lane & 15) + nt * 16;
  int p = j >> 1, e = j & 1;
  int k = kc * 32 + ((lane >> 4) * 16) + p * 2 + e;
  out[idx] = (bf16)W[((size_t)t * 128 + k) * N + n];
}

// ---------------------------------------------------------------------------
__global__ void degree_kernel(const int* __restrict__ dst, float* __restrict__ deg, int E) {
  int i = blockIdx.x * blockDim.x + threadIdx.x;
  if (i < E) atomicAdd(&deg[dst[i]], 1.0f);
}

// ---------------------------------------------------------------------------
// gather + scatter-add: one wave32 per edge, float4 gather + 4x
// global_atomic_add_f32. Optional dropout applied at gather time.
// ---------------------------------------------------------------------------
__global__ void scatter_kernel(const float* __restrict__ h, const int* __restrict__ src,
                               const int* __restrict__ dst, float* __restrict__ agg,
                               int E, int layer) {
  int wave = (int)((blockIdx.x * blockDim.x + threadIdx.x) >> 5);
  int lane = threadIdx.x & 31;
  if (wave >= E) return;
  int s = src[wave], d = dst[wave];
  float4 v = ((const float4*)(h + (size_t)s * 128))[lane];
  float vals[4] = {v.x, v.y, v.z, v.w};
  if (layer) {
    unsigned base = (unsigned)s * 128u + (unsigned)lane * 4u;
#pragma unroll
    for (int i = 0; i < 4; ++i) vals[i] *= drop_scale((unsigned)layer, base + i);
  }
  float* o = agg + (size_t)d * 128 + lane * 4;
  atomicAdd(o + 0, vals[0]);
  atomicAdd(o + 1, vals[1]);
  atomicAdd(o + 2, vals[2]);
  atomicAdd(o + 3, vals[3]);
}

// ---------------------------------------------------------------------------
// Counting sort of node rows by gate type (8 bins). Bucket order is
// atomic-nondeterministic but each row's output value is permutation-invariant.
// ---------------------------------------------------------------------------
__global__ void hist_kernel(const int* __restrict__ types, int* __restrict__ hist, int N) {
  int i = blockIdx.x * blockDim.x + threadIdx.x;
  if (i < N) atomicAdd(&hist[types[i]], 1);
}

__global__ void prefix_kernel(const int* __restrict__ hist, int* __restrict__ off,
                              int* __restrict__ cursor) {
  if (threadIdx.x == 0 && blockIdx.x == 0) {
    int s = 0;
#pragma unroll
    for (int t = 0; t < 8; ++t) { off[t] = s; cursor[t] = s; s += hist[t]; }
    off[8] = s;
  }
}

__global__ void bucket_kernel(const int* __restrict__ types, int* __restrict__ cursor,
                              int* __restrict__ order, int N) {
  int i = blockIdx.x * blockDim.x + threadIdx.x;
  if (i < N) {
    int pos = atomicAdd(&cursor[types[i]], 1);
    order[pos] = i;
  }
}

// ---------------------------------------------------------------------------
// Stage BYTES bytes global -> LDS cooperatively; async-to-LDS when available.
// ---------------------------------------------------------------------------
template <int BYTES>
__device__ __forceinline__ void stage_weights(const bf16* __restrict__ gsrc,
                                              bf16* lds_dst, int tid, int nthreads) {
  const char* g = (const char*)gsrc;
  char* l = (char*)lds_dst;
  const int chunks = BYTES / 16;
#if USE_ASYNC_LDS
#pragma unroll
  for (int i = 0; i < chunks / 256; ++i) {
    int c = tid + i * nthreads;
    size_t off = (size_t)c * 16;
    __builtin_amdgcn_global_load_async_to_lds_b128(
        (as1_v4i*)(uintptr_t)(g + off),
        (as3_v4i*)(unsigned)(uintptr_t)(l + off), 0, 0);
  }
#else
#pragma unroll
  for (int i = 0; i < chunks / 256; ++i) {
    int c = tid + i * nthreads;
    float4 v = ((const float4*)g)[c];
    ((float4*)l)[c] = v;
  }
#endif
}

__device__ __forceinline__ void wait_stage() {
#if USE_ASYNC_LDS
  __builtin_amdgcn_s_wait_asynccnt(0);
#endif
}

// ---------------------------------------------------------------------------
// Single-type GEMM over bucketed rows.
// grid = (maxTilesPerType/8, 8 types); 8 waves/block, one 16-row tile/wave.
// Block stages its type's 32KB bf16 weight block into LDS once, then each
// wave runs exactly 32 WMMAs (K=128, N=128) from ds_load_b128.
// ---------------------------------------------------------------------------
template <bool ACT>
__global__ __launch_bounds__(256) void gemm_sorted_kernel(
    const float* __restrict__ A, const float* __restrict__ deg,
    const int* __restrict__ order, const int* __restrict__ off,
    const bf16* __restrict__ Wp, const float* __restrict__ bias,
    float* __restrict__ out) {
  __shared__ bf16 WB[4 * 8 * 512];                 // 32KB: one type's weights

  const int ty    = blockIdx.y;
  const int start = off[ty];
  const int cnt   = off[ty + 1] - start;
  const int tilesInType = (cnt + 15) >> 4;
  if ((int)blockIdx.x * 8 >= tilesInType) return;  // uniform: whole block idle

  const int lane = threadIdx.x & 31;
  const int tileT = blockIdx.x * 8 + (threadIdx.x >> 5);

  stage_weights<32768>(Wp + (size_t)ty * 16384, WB, threadIdx.x, 256);
  wait_stage();
  __syncthreads();

  const bool waveValid = tileT < tilesInType;      // wave-uniform
  const int tileC = waveValid ? tileT : 0;
  const int m = lane & 15;
  const int slot = tileC * 16 + m;
  const int slotC = slot < cnt ? slot : (cnt - 1); // clamp (cnt>=1 here)
  const int row = order[start + slotC];

  const float scale = 1.0f / fmaxf(deg[row], 1.0f);
  const float* arow = A + (size_t)row * 128;
  const int kbase = (lane >> 4) * 16;

  v8f acc[8];
  v8f zf = {};
#pragma unroll
  for (int nt = 0; nt < 8; ++nt) acc[nt] = zf;

#pragma unroll
  for (int kc = 0; kc < 4; ++kc) {
    // A fragment for this K-chunk (row cached in L0 across chunks)
    v16bf av;
#pragma unroll
    for (int p = 0; p < 8; ++p) {
      int k = kc * 32 + kbase + p * 2;
      float2 x = *(const float2*)(arow + k);
      av[2 * p]     = (bf16)(x.x * scale);
      av[2 * p + 1] = (bf16)(x.y * scale);
    }
#pragma unroll
    for (int nt = 0; nt < 8; ++nt) {
      v16bf b = *(const v16bf*)(&WB[0] + ((size_t)kc * 8 + nt) * 512 + lane * 16);
      acc[nt] = __builtin_amdgcn_wmma_f32_16x16x32_bf16(
          false, av, false, b, (short)0, acc[nt], false, false);
    }
  }

  if (!waveValid) return;

  // Epilogue. C/D layout: VGPR r, lane l -> M = r + 8*(l>>4), N = l&15
  const int n0 = lane & 15;
#pragma unroll
  for (int r = 0; r < 8; ++r) {
    int mslot = tileC * 16 + r + ((lane >> 4) << 3);
    bool rowOk = mslot < cnt;
    int orow = order[start + (rowOk ? mslot : (cnt - 1))];
#pragma unroll
    for (int nt = 0; nt < 8; ++nt) {
      int col = nt * 16 + n0;
      float v = acc[nt][r] + bias[ty * 128 + col];
      if (ACT) v = fmaxf(v, 0.0f);
      if (rowOk) out[(size_t)orow * 128 + col] = v;
    }
  }
}

// ---------------------------------------------------------------------------
// fc GEMM: [nRows,128] @ [128, NT*16] + b. Tiny weight -> register path.
// ---------------------------------------------------------------------------
template <int NT>
__global__ __launch_bounds__(256) void gemm_fc_kernel(
    const float* __restrict__ A, const bf16* __restrict__ Wp,
    const float* __restrict__ bias, float* __restrict__ out, int nRows) {
  const int lane  = threadIdx.x & 31;
  const int tile  = blockIdx.x * (blockDim.x >> 5) + (threadIdx.x >> 5);
  const int tiles = nRows >> 4;
  if (tile >= tiles) return;                    // wave-uniform
  const int rowBase = tile << 4;
  const int row = rowBase + (lane & 15);

  const float* arow = A + (size_t)row * 128;
  const int kbase = (lane >> 4) * 16;

  v8f acc[NT];
  v8f zf = {};
#pragma unroll
  for (int nt = 0; nt < NT; ++nt) acc[nt] = zf;

#pragma unroll
  for (int kc = 0; kc < 4; ++kc) {
    v16bf av;
#pragma unroll
    for (int p = 0; p < 8; ++p) {
      int k = kc * 32 + kbase + p * 2;
      float2 x = *(const float2*)(arow + k);
      av[2 * p]     = (bf16)x.x;
      av[2 * p + 1] = (bf16)x.y;
    }
#pragma unroll
    for (int nt = 0; nt < NT; ++nt) {
      v16bf b = *(const v16bf*)(Wp + ((size_t)kc * NT + nt) * 512 + lane * 16);
      acc[nt] = __builtin_amdgcn_wmma_f32_16x16x32_bf16(
          false, av, false, b, (short)0, acc[nt], false, false);
    }
  }

  const int n0 = lane & 15;
  const int DOUT = NT * 16;
#pragma unroll
  for (int r = 0; r < 8; ++r) {
    int orow = rowBase + r + ((lane >> 4) << 3);
#pragma unroll
    for (int nt = 0; nt < NT; ++nt) {
      int col = nt * 16 + n0;
      out[(size_t)orow * DOUT + col] = acc[nt][r] + bias[col];
    }
  }
}

// ---------------------------------------------------------------------------
extern "C" void kernel_launch(void* const* d_in, const int* in_sizes, int n_in,
                              void* d_out, int out_size, void* d_ws, size_t ws_size,
                              hipStream_t stream) {
  const float* features = (const float*)d_in[0];
  const int*   src0     = (const int*)d_in[1];
  const int*   dst0     = (const int*)d_in[2];
  const int*   types1   = (const int*)d_in[3];
  const int*   src1     = (const int*)d_in[4];
  const int*   dst1     = (const int*)d_in[5];
  const int*   types2   = (const int*)d_in[6];
  const int*   src2     = (const int*)d_in[7];
  const int*   dst2     = (const int*)d_in[8];
  const int*   types3   = (const int*)d_in[9];
  const float* convW    = (const float*)d_in[10];
  const float* convb    = (const float*)d_in[11];
  const float* fcW      = (const float*)d_in[12];
  const float* fcb      = (const float*)d_in[13];

  const int E0 = in_sizes[1], N1 = in_sizes[3];
  const int E1 = in_sizes[4], N2 = in_sizes[6];
  const int E2 = in_sizes[7], N3 = in_sizes[9];
  const int T  = in_sizes[10] / (128 * 128);   // 8 gate types
  const int DOUT = in_sizes[13];               // 64

  auto align256 = [](size_t x) { return (x + 255) & ~(size_t)255; };
  char* p = (char*)d_ws;
  float* agg    = (float*)p;  p += align256((size_t)N1 * 128 * sizeof(float));
  float* hbuf   = (float*)p;  p += align256((size_t)N1 * 128 * sizeof(float));
  float* deg    = (float*)p;  p += align256((size_t)N1 * sizeof(float));
  bf16*  packW  = (bf16*)p;   p += align256((size_t)T * 128 * 128 * sizeof(bf16));
  bf16*  packFc = (bf16*)p;   p += align256((size_t)128 * DOUT * sizeof(bf16));
  int*   order  = (int*)p;    p += align256((size_t)N1 * sizeof(int));
  int*   hist   = (int*)p;    p += align256(16 * sizeof(int));
  int*   off    = (int*)p;    p += align256(16 * sizeof(int));
  int*   cursor = (int*)p;

  // Pack weights into bf16 WMMA-tile layout (cheap, per-call, deterministic).
  { int tot = T * 4 * 8 * 512;
    pack_w_kernel<<<(tot + 255) / 256, 256, 0, stream>>>(convW, packW, T, 8); }
  { int nt = DOUT / 16; int tot = 4 * nt * 512;
    pack_w_kernel<<<(tot + 255) / 256, 256, 0, stream>>>(fcW, packFc, 1, nt); }

  // Per layer: mean-aggregate over edges, bucket rows by type, single-type GEMM.
  struct Layer {
    const float* h; const int* s; const int* d; const int* ty;
    int E; int n; int li;
  };
  const Layer layers[3] = {
    { features, src0, dst0, types1, E0, N1, 0 },
    { hbuf,     src1, dst1, types2, E1, N2, 1 },
    { hbuf,     src2, dst2, types3, E2, N3, 2 },
  };

  for (int i = 0; i < 3; ++i) {
    const Layer& L = layers[i];
    (void)hipMemsetAsync(agg, 0, (size_t)L.n * 128 * sizeof(float), stream);
    (void)hipMemsetAsync(deg, 0, (size_t)L.n * sizeof(float), stream);
    (void)hipMemsetAsync(hist, 0, 8 * sizeof(int), stream);
    degree_kernel<<<(L.E + 255) / 256, 256, 0, stream>>>(L.d, deg, L.E);
    scatter_kernel<<<(L.E + 7) / 8, 256, 0, stream>>>(L.h, L.s, L.d, agg, L.E, L.li);
    hist_kernel<<<(L.n + 255) / 256, 256, 0, stream>>>(L.ty, hist, L.n);
    prefix_kernel<<<1, 32, 0, stream>>>(hist, off, cursor);
    bucket_kernel<<<(L.n + 255) / 256, 256, 0, stream>>>(L.ty, cursor, order, L.n);
    int tiles = L.n / 16;
    dim3 grid((tiles + 7) / 8, 8);
    if (i != 2) {
      gemm_sorted_kernel<true><<<grid, 256, 0, stream>>>(
          agg, deg, order, off, packW, convb, hbuf);
    } else {
      gemm_sorted_kernel<false><<<grid, 256, 0, stream>>>(
          agg, deg, order, off, packW, convb, hbuf);
    }
  }

  // fc: [N3,128] @ [128,64] + b
  { int tiles = N3 / 16;
    gemm_fc_kernel<4><<<(tiles + 7) / 8, 256, 0, stream>>>(
        hbuf, packFc, fcb, (float*)d_out, N3); }
}
